// DKAJSummary_88098369176476
// MI455X (gfx1250) — compile-verified, matrix operand order checked by reading, not run on hardware
//
#include <hip/hip_runtime.h>
#include <hip/hip_bf16.h>

#define B_N 8192
#define E_N 4096
#define D_N 256
#define T_N 128
#define R_N 4
#define K_N 64
#define QCHUNK 1024
#define TAU_SQ 4.0f
#define EPSV 1e-12f

typedef float     v2f  __attribute__((ext_vector_type(2)));
typedef float     v8f  __attribute__((ext_vector_type(8)));
typedef _Float16  v16h __attribute__((ext_vector_type(16)));

#if defined(__has_builtin)
#if __has_builtin(__builtin_amdgcn_wmma_f32_16x16x4_f32)
#define HAVE_WMMA_F32X4 1
#endif
#endif

__device__ __forceinline__ float clampf(float v) {
    return fminf(fmaxf(v, EPSV), 1.0f - EPSV);
}

// ---------------------------------------------------------------------------
// Kernel 1: exp() + reverse (suffix) cumsum over T for exemplars + baseline.
// Grid: E_N+1 blocks of T_N threads. Block E_N handles the baseline arrays.
// ---------------------------------------------------------------------------
__global__ void dkaj_prep(const float* __restrict__ log_ev,
                          const float* __restrict__ log_cen,
                          const float* __restrict__ log_b_ev,
                          const float* __restrict__ log_b_cen,
                          float* __restrict__ ev,        // [E,T,R]
                          float* __restrict__ at_risk,   // [E,T]
                          float* __restrict__ b_ev,      // [T,R]
                          float* __restrict__ b_at_risk) // [T]
{
    __shared__ float s[T_N];
    const int t = threadIdx.x;
    const int e = blockIdx.x;
    float sum;
    if (e < E_N) {
        const size_t et = (size_t)e * T_N + t;
        float4 v = *(const float4*)(log_ev + et * R_N);
        float4 o;
        o.x = __expf(v.x); o.y = __expf(v.y); o.z = __expf(v.z); o.w = __expf(v.w);
        *(float4*)(ev + et * R_N) = o;
        sum = o.x + o.y + o.z + o.w + __expf(log_cen[et]);
    } else {
        float4 v = *(const float4*)(log_b_ev + t * R_N);
        float4 o;
        o.x = __expf(v.x); o.y = __expf(v.y); o.z = __expf(v.z); o.w = __expf(v.w);
        *(float4*)(b_ev + t * R_N) = o;
        sum = o.x + o.y + o.z + o.w + __expf(log_b_cen[t]);
    }
    s[t] = sum;
    __syncthreads();
    // inclusive suffix scan (== flip-cumsum-flip)
    for (int d = 1; d < T_N; d <<= 1) {
        float cur = s[t];
        float add = (t + d < T_N) ? s[t + d] : 0.0f;
        __syncthreads();
        s[t] = cur + add;
        __syncthreads();
    }
    if (e < E_N) at_risk[(size_t)e * T_N + t] = s[t];
    else         b_at_risk[t] = s[t];
}

// ---------------------------------------------------------------------------
// Kernel 2: row squared norms of x and ex. Grid: B_N+E_N blocks, 64 threads.
// ---------------------------------------------------------------------------
__global__ void dkaj_norms(const float* __restrict__ x, const float* __restrict__ ex,
                           float* __restrict__ xx, float* __restrict__ ee)
{
    __shared__ float red[64];
    const int tid = threadIdx.x;
    const int row = blockIdx.x;
    const float* p = (row < B_N) ? (x + (size_t)row * D_N)
                                 : (ex + (size_t)(row - B_N) * D_N);
    float4 v = *(const float4*)(p + tid * 4);
    red[tid] = v.x * v.x + v.y * v.y + v.z * v.z + v.w * v.w;
    __syncthreads();
    for (int d = 32; d > 0; d >>= 1) {
        if (tid < d) red[tid] += red[tid + d];
        __syncthreads();
    }
    if (tid == 0) {
        if (row < B_N) xx[row] = red[0];
        else           ee[row - B_N] = red[0];
    }
}

// ---------------------------------------------------------------------------
// Kernel 3: WMMA distance tiles -> gated exp weights W[qchunk, E].
// Block = 256 threads = 8 wave32; each wave computes FOUR adjacent 16x16
// e-tiles reusing the A (query) fragment: per k-step 1 A load + 4 B loads
// feed 4 v_wmma_f32_16x16x4_f32 (6.4 FLOP/byte vs 4 with one tile).
// A(16x4): lanes 0-15 hold M=lane, K={0,1}; lanes 16-31 K={2,3}.
// B(4x16) mirrors it with N=lane&15.
// ---------------------------------------------------------------------------
__global__ void __launch_bounds__(256)
dkaj_dist(const float* __restrict__ x, const float* __restrict__ ex,
          const float* __restrict__ xx, const float* __restrict__ ee,
          float* __restrict__ W, int qbase)
{
    const int wave  = threadIdx.x >> 5;
    const int lane  = threadIdx.x & 31;
    const int q0    = qbase + blockIdx.x * 16;
    const int e0    = blockIdx.y * 512 + wave * 64;   // 4 tiles of 16
    const int row   = lane & 15;
    const int khalf = lane >> 4;

    v8f acc0 = {}, acc1 = {}, acc2 = {}, acc3 = {};
#if HAVE_WMMA_F32X4
    const float* xp  = x  + (size_t)(q0 + row) * D_N + 2 * khalf;
    const float* ep0 = ex + (size_t)(e0 +  0 + row) * D_N + 2 * khalf;
    const float* ep1 = ex + (size_t)(e0 + 16 + row) * D_N + 2 * khalf;
    const float* ep2 = ex + (size_t)(e0 + 32 + row) * D_N + 2 * khalf;
    const float* ep3 = ex + (size_t)(e0 + 48 + row) * D_N + 2 * khalf;
    #pragma unroll
    for (int k = 0; k < D_N; k += 4) {
        v2f a  = *(const v2f*)(xp  + k);
        v2f b0 = *(const v2f*)(ep0 + k);
        v2f b1 = *(const v2f*)(ep1 + k);
        v2f b2 = *(const v2f*)(ep2 + k);
        v2f b3 = *(const v2f*)(ep3 + k);
        acc0 = __builtin_amdgcn_wmma_f32_16x16x4_f32(false, a, false, b0, (short)0, acc0, false, false);
        acc1 = __builtin_amdgcn_wmma_f32_16x16x4_f32(false, a, false, b1, (short)0, acc1, false, false);
        acc2 = __builtin_amdgcn_wmma_f32_16x16x4_f32(false, a, false, b2, (short)0, acc2, false, false);
        acc3 = __builtin_amdgcn_wmma_f32_16x16x4_f32(false, a, false, b3, (short)0, acc3, false, false);
    }
#else
    // f16 fallback (16x16x32): 16-bit A 16x32 layout per ISA 7.12.2.
    const float* xp  = x  + (size_t)(q0 + row) * D_N;
    const float* ep0 = ex + (size_t)(e0 +  0 + row) * D_N;
    const float* ep1 = ex + (size_t)(e0 + 16 + row) * D_N;
    const float* ep2 = ex + (size_t)(e0 + 32 + row) * D_N;
    const float* ep3 = ex + (size_t)(e0 + 48 + row) * D_N;
    for (int k0 = 0; k0 < D_N; k0 += 32) {
        v16h a, b0, b1, b2, b3;
        #pragma unroll
        for (int p = 0; p < 8; ++p) {
            const int kk = k0 + ((p >= 4) ? 16 : 0) + ((p & 3) * 2) + 8 * khalf;
            float2 av  = *(const float2*)(xp  + kk);
            float2 bv0 = *(const float2*)(ep0 + kk);
            float2 bv1 = *(const float2*)(ep1 + kk);
            float2 bv2 = *(const float2*)(ep2 + kk);
            float2 bv3 = *(const float2*)(ep3 + kk);
            a [2*p] = (_Float16)av.x;  a [2*p+1] = (_Float16)av.y;
            b0[2*p] = (_Float16)bv0.x; b0[2*p+1] = (_Float16)bv0.y;
            b1[2*p] = (_Float16)bv1.x; b1[2*p+1] = (_Float16)bv1.y;
            b2[2*p] = (_Float16)bv2.x; b2[2*p+1] = (_Float16)bv2.y;
            b3[2*p] = (_Float16)bv3.x; b3[2*p+1] = (_Float16)bv3.y;
        }
        acc0 = __builtin_amdgcn_wmma_f32_16x16x32_f16(false, a, false, b0, (short)0, acc0, false, false);
        acc1 = __builtin_amdgcn_wmma_f32_16x16x32_f16(false, a, false, b1, (short)0, acc1, false, false);
        acc2 = __builtin_amdgcn_wmma_f32_16x16x32_f16(false, a, false, b2, (short)0, acc2, false, false);
        acc3 = __builtin_amdgcn_wmma_f32_16x16x32_f16(false, a, false, b3, (short)0, acc3, false, false);
    }
#endif
    // C/D layout: vgpr v -> M = v + 8*khalf, N = lane&15
    const int n = row;
    float* wbase = W + (size_t)(blockIdx.x * 16) * E_N + e0 + n;
    float xq[8];
    #pragma unroll
    for (int v = 0; v < 8; ++v) xq[v] = xx[q0 + v + 8 * khalf];

    auto emit = [&](const v8f& acc, int eoff) {
        const float een = ee[e0 + eoff + n];
        float* wout = wbase + eoff;
        #pragma unroll
        for (int v = 0; v < 8; ++v) {
            const int m = v + 8 * khalf;
            float d2 = xq[v] + een - 2.0f * acc[v];
            d2 = fmaxf(d2, 0.0f);
            wout[(size_t)m * E_N] = (d2 <= TAU_SQ) ? __expf(-d2) : 0.0f;
        }
    };
    emit(acc0, 0); emit(acc1, 16); emit(acc2, 32); emit(acc3, 48);
}

// ---------------------------------------------------------------------------
// Kernel 4: exact top-64 per query + sparse accumulation + clipped outputs.
// Block per query, 256 threads, 16 weights/thread held in registers.
// ---------------------------------------------------------------------------
__global__ void __launch_bounds__(256)
dkaj_select(const float* __restrict__ W,          // [QCHUNK, E]
            const float* __restrict__ at_risk,    // [E,T]
            const float* __restrict__ ev,         // [E,T,R]
            const float* __restrict__ b_ev,       // [T,R]
            const float* __restrict__ b_at_risk,  // [T]
            float* __restrict__ out, int qbase)
{
    __shared__ float s_redv[8];
    __shared__ int   s_redi[8];
    __shared__ int   s_besti;
    __shared__ float s_selw[K_N];
    __shared__ int   s_sele[K_N];
    __shared__ float s_acc[5 * 256];

    const int tid  = threadIdx.x;
    const int lane = tid & 31;
    const int wid  = tid >> 5;
    const float* wrow = W + (size_t)blockIdx.x * E_N;

    float vals[16];
    #pragma unroll
    for (int i = 0; i < 16; ++i) vals[i] = wrow[i * 256 + tid];

    float lmax = vals[0]; int lidx = tid;
    #pragma unroll
    for (int i = 1; i < 16; ++i)
        if (vals[i] > lmax) { lmax = vals[i]; lidx = i * 256 + tid; }

    for (int k = 0; k < K_N; ++k) {
        float v = lmax; int id = lidx;
        #pragma unroll
        for (int off = 16; off >= 1; off >>= 1) {
            float ov = __shfl_xor(v, off, 32);
            int   oi = __shfl_xor(id, off, 32);
            if (ov > v || (ov == v && oi < id)) { v = ov; id = oi; }
        }
        if (lane == 0) { s_redv[wid] = v; s_redi[wid] = id; }
        __syncthreads();
        if (tid == 0) {
            float bv = s_redv[0]; int bi = s_redi[0];
            for (int w2 = 1; w2 < 8; ++w2) {
                float ov = s_redv[w2]; int oi = s_redi[w2];
                if (ov > bv || (ov == bv && oi < bi)) { bv = ov; bi = oi; }
            }
            s_besti = bi; s_selw[k] = bv; s_sele[k] = bi;
        }
        __syncthreads();
        const int bi = s_besti;
        if ((bi & 255) == tid) {          // owner removes winner, rescans
            vals[bi >> 8] = -1.0f;
            lmax = vals[0]; lidx = tid;
            #pragma unroll
            for (int i = 1; i < 16; ++i)
                if (vals[i] > lmax) { lmax = vals[i]; lidx = i * 256 + tid; }
        }
        __syncthreads();
    }

    // sparse accumulation: 2 threads per time bin (k-ranges 0-31 / 32-63)
    const int t    = tid & 127;
    const int half = tid >> 7;
    float dsum = 0.0f, n0 = 0.0f, n1 = 0.0f, n2 = 0.0f, n3 = 0.0f;
    for (int k = half * 32; k < half * 32 + 32; ++k) {
        const float wgt = s_selw[k];
        const int   e   = s_sele[k];
        const size_t et = (size_t)e * T_N + t;
        dsum = fmaf(wgt, at_risk[et], dsum);
        const float4 evv = *(const float4*)(ev + et * R_N);
        n0 = fmaf(wgt, evv.x, n0);
        n1 = fmaf(wgt, evv.y, n1);
        n2 = fmaf(wgt, evv.z, n2);
        n3 = fmaf(wgt, evv.w, n3);
    }
    s_acc[tid]        = dsum;
    s_acc[256  + tid] = n0;
    s_acc[512  + tid] = n1;
    s_acc[768  + tid] = n2;
    s_acc[1024 + tid] = n3;
    __syncthreads();
    if (tid < 128) {
        const float dd = s_acc[tid] + s_acc[tid + 128] + b_at_risk[t] + EPSV;
        const float m0 = s_acc[256  + tid] + s_acc[256  + tid + 128] + b_ev[t * 4 + 0];
        const float m1 = s_acc[512  + tid] + s_acc[512  + tid + 128] + b_ev[t * 4 + 1];
        const float m2 = s_acc[768  + tid] + s_acc[768  + tid + 128] + b_ev[t * 4 + 2];
        const float m3 = s_acc[1024 + tid] + s_acc[1024 + tid + 128] + b_ev[t * 4 + 3];
        const float inv = 1.0f / dd;
        const size_t b  = (size_t)qbase + blockIdx.x;
        const size_t BT = (size_t)B_N * T_N;
        out[0 * BT + b * T_N + t] = clampf(m0 * inv);
        out[1 * BT + b * T_N + t] = clampf(m1 * inv);
        out[2 * BT + b * T_N + t] = clampf(m2 * inv);
        out[3 * BT + b * T_N + t] = clampf(m3 * inv);
        out[4 * BT + b * T_N + t] = clampf((m0 + m1 + m2 + m3) * inv);
    }
}

// ---------------------------------------------------------------------------
extern "C" void kernel_launch(void* const* d_in, const int* in_sizes, int n_in,
                              void* d_out, int out_size, void* d_ws, size_t ws_size,
                              hipStream_t stream)
{
    const float* x         = (const float*)d_in[0];
    const float* ex        = (const float*)d_in[1];
    const float* log_ev    = (const float*)d_in[2];
    const float* log_cen   = (const float*)d_in[3];
    const float* log_b_ev  = (const float*)d_in[4];
    const float* log_b_cen = (const float*)d_in[5];
    float* out = (float*)d_out;

    float* ws     = (float*)d_ws;
    float* ws_ev  = ws;                                // E*T*R
    float* ws_ar  = ws_ev + (size_t)E_N * T_N * R_N;   // E*T
    float* ws_bev = ws_ar + (size_t)E_N * T_N;         // T*R
    float* ws_bar = ws_bev + T_N * R_N;                // T
    float* ws_xx  = ws_bar + T_N;                      // B
    float* ws_ee  = ws_xx + B_N;                       // E
    float* ws_W   = ws_ee + E_N;                       // QCHUNK*E  (reused per chunk)

    dkaj_prep<<<E_N + 1, T_N, 0, stream>>>(log_ev, log_cen, log_b_ev, log_b_cen,
                                           ws_ev, ws_ar, ws_bev, ws_bar);
    dkaj_norms<<<B_N + E_N, 64, 0, stream>>>(x, ex, ws_xx, ws_ee);

    for (int qb = 0; qb < B_N; qb += QCHUNK) {
        dkaj_dist<<<dim3(QCHUNK / 16, E_N / 512), 256, 0, stream>>>(
            x, ex, ws_xx, ws_ee, ws_W, qb);
        dkaj_select<<<QCHUNK, 256, 0, stream>>>(
            ws_W, ws_ar, ws_ev, ws_bev, ws_bar, out, qb);
    }
}
